// VRNN_66365834658104
// MI455X (gfx1250) — compile-verified, matrix operand order checked by reading, not run on hardware
//
#include <hip/hip_runtime.h>
#include <hip/hip_bf16.h>
#include <math.h>

// ---------------- problem constants ----------------
#define BB 32
#define TT 128
#define XX 512
#define HH 1024
#define ZZ 256
#define VV 32000

typedef __attribute__((ext_vector_type(16))) __bf16    v16bf;
typedef __attribute__((ext_vector_type(8)))  float     v8f;
typedef __attribute__((ext_vector_type(4)))  float     v4f;
typedef __attribute__((ext_vector_type(4)))  unsigned  v4u;

// activation selectors
#define ACT_NONE 0
#define ACT_RELU 1
#define ACT_SIGMOID 2
#define ACT_SOFTPLUS 3

__device__ __forceinline__ float actf(float x, int act) {
  switch (act) {
    case ACT_RELU:     return x > 0.f ? x : 0.f;
    case ACT_SIGMOID:  return 1.f / (1.f + __expf(-x));
    case ACT_SOFTPLUS: return (x > 20.f) ? x : log1pf(__expf(x));
    default:           return x;
  }
}

// deterministic hash-based standard normal (stand-in for jax PRNG)
__device__ __forceinline__ unsigned pcg(unsigned v) {
  v = v * 747796405u + 2891336453u;
  unsigned w = ((v >> ((v >> 28u) + 4u)) ^ v) * 277803737u;
  return (w >> 22u) ^ w;
}
__device__ __forceinline__ float gaussian(unsigned s) {
  unsigned u1 = pcg(s);
  unsigned u2 = pcg(s ^ 0x9E3779B9u);
  float f1 = ((float)u1 + 1.0f) * 2.3283064365386963e-10f;  // (0,1]
  float f2 = (float)u2 * 2.3283064365386963e-10f;
  return sqrtf(-2.0f * __logf(f1)) * __cosf(6.283185307179586f * f2);
}

// ---------------------------------------------------------------------------
// Generic fused GEMM: Y[M x N] = act( X[M x Ktot] @ W[N x Ktot]^T + bias )
//   * MREP: M-blocks (of 32 rows) per workgroup. The B (weight) fragment is
//     loaded once per k-step and reused by 2*MREP WMMAs -> weight traffic,
//     the dominant term, is divided by MREP (fc uses MREP=4: 128x64 tile).
//   * X given as two fp32 sources (X1 width K1, row stride ldX1; optional X2
//     width K2 contiguous) so concat([a, h]) never materializes. Two uniform
//     K passes -> no per-lane branch, no EXEC churn.
//   * Per-column activation split: col < nsplit -> act1 else act2, letting
//     [mean; std] head pairs fuse into one launch (per-lane select only;
//     WMMA itself always runs with full EXEC).
//   * Fragments per CDNA5 ISA 7.12.2 16-bit layouts: per lane two contiguous
//     8-element K chunks at +8*hsel / +16+8*hsel -> b128 loads throughout.
// ---------------------------------------------------------------------------
union Frag { v16bf v; v4u u2[2]; };

template <int MREP>
__global__ __launch_bounds__(128)
void wmma_gemm(const float* __restrict__ X1, int K1, int ldX1,
               const float* __restrict__ X2, int K2,
               const __bf16* __restrict__ W, const float* __restrict__ bias,
               float* __restrict__ Y, int N, int act1, int act2, int nsplit)
{
  const int lane = threadIdx.x & 31;
  const int wave = threadIdx.x >> 5;
  const int r    = lane & 15;
  const int hsel = lane >> 4;                      // which K half-group
  const int col  = blockIdx.x * 64 + wave * 16 + r;
  const int rowBase0 = blockIdx.y * (32 * MREP);
  const int Ktot = K1 + K2;

  v8f acc[MREP][2];
#pragma unroll
  for (int m = 0; m < MREP; ++m) { acc[m][0] = (v8f){}; acc[m][1] = (v8f){}; }

  const __bf16* wrow = W + (size_t)col * Ktot + 8 * hsel;

  auto kpass = [&](const float* __restrict__ X, int ldX, int Kc,
                   const __bf16* __restrict__ wp) {
    const float* xb = X + 8 * hsel;
    for (int k0 = 0; k0 < Kc; k0 += 32) {
      Frag b;
      b.u2[0] = *(const v4u*)(wp + k0);            // two 16B weight chunks
      b.u2[1] = *(const v4u*)(wp + k0 + 16);
#pragma unroll
      for (int m = 0; m < MREP; ++m) {
        const float* xr0 = xb + (size_t)(rowBase0 + m * 32 + r) * ldX + k0;
        const float* xr1 = xb + (size_t)(rowBase0 + m * 32 + 16 + r) * ldX + k0;
        Frag a0, a1;
        v4f x00 = *(const v4f*)(xr0);
        v4f x01 = *(const v4f*)(xr0 + 4);
        v4f x02 = *(const v4f*)(xr0 + 16);
        v4f x03 = *(const v4f*)(xr0 + 20);
        v4f x10 = *(const v4f*)(xr1);
        v4f x11 = *(const v4f*)(xr1 + 4);
        v4f x12 = *(const v4f*)(xr1 + 16);
        v4f x13 = *(const v4f*)(xr1 + 20);
#pragma unroll
        for (int e = 0; e < 4; ++e) {
          a0.v[e]      = (__bf16)x00[e];
          a0.v[4 + e]  = (__bf16)x01[e];
          a0.v[8 + e]  = (__bf16)x02[e];
          a0.v[12 + e] = (__bf16)x03[e];
          a1.v[e]      = (__bf16)x10[e];
          a1.v[4 + e]  = (__bf16)x11[e];
          a1.v[8 + e]  = (__bf16)x12[e];
          a1.v[12 + e] = (__bf16)x13[e];
        }
        acc[m][0] = __builtin_amdgcn_wmma_f32_16x16x32_bf16(
            false, a0.v, false, b.v, (short)0, acc[m][0], false, false);
        acc[m][1] = __builtin_amdgcn_wmma_f32_16x16x32_bf16(
            false, a1.v, false, b.v, (short)0, acc[m][1], false, false);
      }
    }
  };

  kpass(X1, ldX1, K1, wrow);
  if (K2 > 0) kpass(X2, K2, K2, wrow + K1);   // uniform branch

  const float bv = bias ? bias[col] : 0.f;
  const int   act = (col < nsplit) ? act1 : act2;
#pragma unroll
  for (int m = 0; m < MREP; ++m) {
#pragma unroll
    for (int i = 0; i < 8; ++i) {
      const int m0 = rowBase0 + m * 32 + i + 8 * hsel;   // C/D: VGPR i -> M
      const int m1 = rowBase0 + m * 32 + 16 + i + 8 * hsel;
      Y[(size_t)m0 * N + col] = actf(acc[m][0][i] + bv, act);
      Y[(size_t)m1 * N + col] = actf(acc[m][1][i] + bv, act);
    }
  }
}

// ---------------- small helper kernels ----------------
__global__ void cvt_bf16_kernel(const float* __restrict__ s,
                                __bf16* __restrict__ d, int n) {
  int i = blockIdx.x * 256 + threadIdx.x;
  if (i < n) d[i] = (__bf16)s[i];
}

__global__ void copyf_kernel(const float* __restrict__ s,
                             float* __restrict__ d, int n) {
  int i = blockIdx.x * 256 + threadIdx.x;
  if (i < n) d[i] = s[i];
}

__global__ void zero2_kernel(float* o) {
  if (threadIdx.x < 2) o[threadIdx.x] = 0.f;
}

// z = eps * es + em ; em/es packed as [B x (Z|Z)]
__global__ void sample_z_kernel(const float* __restrict__ emes,
                                float* __restrict__ z, int t) {
  int i = blockIdx.x * 256 + threadIdx.x;
  if (i < BB * ZZ) {
    int b = i / ZZ, j = i % ZZ;
    float em = emes[b * 2 * ZZ + j];
    float es = emes[b * 2 * ZZ + ZZ + j];
    float eps = gaussian(0x10000001u + (unsigned)(t * BB * ZZ + i));
    z[i] = eps * es + em;
  }
}

// pred = dm + exp(ds/2)*eps ; dm/ds packed as [B x (X|X)] ; -> preds[t]
__global__ void pred_kernel(const float* __restrict__ dmds,
                            float* __restrict__ preds, int t) {
  int i = blockIdx.x * 256 + threadIdx.x;
  if (i < BB * XX) {
    int b = i / XX, j = i % XX;
    float dm = dmds[b * 2 * XX + j];
    float ds = dmds[b * 2 * XX + XX + j];
    float eps = gaussian(0x20000002u + (unsigned)(t * BB * XX + i));
    preds[(size_t)t * BB * XX + i] = dm + __expf(0.5f * ds) * eps;
  }
}

// kld += 0.5 * sum( 2log(ps) - 2log(es) + (es^2+(em-pm)^2)/ps^2 - 1 )
__global__ void kld_kernel(const float* __restrict__ emes,
                           const float* __restrict__ pmps,
                           float* __restrict__ out) {
  __shared__ float red[256];
  int i = blockIdx.x * 256 + threadIdx.x;
  float v = 0.f;
  if (i < BB * ZZ) {
    int b = i / ZZ, j = i % ZZ;
    float em = emes[b * 2 * ZZ + j], es = emes[b * 2 * ZZ + ZZ + j];
    float pm = pmps[b * 2 * ZZ + j], ps = pmps[b * 2 * ZZ + ZZ + j];
    float d = em - pm;
    v = 2.f * __logf(ps) - 2.f * __logf(es) + (es * es + d * d) / (ps * ps) - 1.f;
  }
  red[threadIdx.x] = v;
  __syncthreads();
  for (int s = 128; s > 0; s >>= 1) {
    if (threadIdx.x < s) red[threadIdx.x] += red[threadIdx.x + s];
    __syncthreads();
  }
  if (threadIdx.x == 0) atomicAdd(out, 0.5f * red[0]);
}

// nll -= sum( x*log(dm) + (1-x)*log(1-dm) ) ; x_t strided [B, T*X]
__global__ void nll_kernel(const float* __restrict__ xbase,
                           const float* __restrict__ dmds,
                           float* __restrict__ out) {
  __shared__ float red[256];
  int i = blockIdx.x * 256 + threadIdx.x;
  float v = 0.f;
  if (i < BB * XX) {
    int b = i / XX, j = i % XX;
    float xt = xbase[(size_t)b * TT * XX + j];
    float m = dmds[b * 2 * XX + j];
    v = xt * __logf(m) + (1.f - xt) * __logf(1.f - m);
  }
  red[threadIdx.x] = v;
  __syncthreads();
  for (int s = 128; s > 0; s >>= 1) {
    if (threadIdx.x < s) red[threadIdx.x] += red[threadIdx.x + s];
    __syncthreads();
  }
  if (threadIdx.x == 0) atomicAdd(out, -red[0]);
}

// GRU gates (torch order r, z, n; no bias)
__global__ void gru_kernel(const float* __restrict__ gi, const float* __restrict__ gh,
                           const float* __restrict__ hb, float* __restrict__ hn) {
  int i = blockIdx.x * 256 + threadIdx.x;
  if (i < BB * HH) {
    int b = i / HH, j = i % HH;
    const float* gib = gi + (size_t)b * 3 * HH;
    const float* ghb = gh + (size_t)b * 3 * HH;
    float r  = 1.f / (1.f + __expf(-(gib[j] + ghb[j])));
    float zg = 1.f / (1.f + __expf(-(gib[HH + j] + ghb[HH + j])));
    float n  = tanhf(gib[2 * HH + j] + r * ghb[2 * HH + j]);
    hn[i] = (1.f - zg) * n + zg * hb[i];
  }
}

// ---------------------------------------------------------------------------
extern "C" void kernel_launch(void* const* d_in, const int* in_sizes, int n_in,
                              void* d_out, int out_size, void* d_ws, size_t ws_size,
                              hipStream_t stream) {
  (void)in_sizes; (void)n_in; (void)out_size;
  const float* x             = (const float*)d_in[0];
  const float* initial_state = (const float*)d_in[3];
  const float* phi_x_w1 = (const float*)d_in[4];  const float* phi_x_b1 = (const float*)d_in[5];
  const float* phi_x_w2 = (const float*)d_in[6];  const float* phi_x_b2 = (const float*)d_in[7];
  const float* phi_z_w  = (const float*)d_in[8];  const float* phi_z_b  = (const float*)d_in[9];
  const float* enc_w1   = (const float*)d_in[10]; const float* enc_b1   = (const float*)d_in[11];
  const float* enc_w2   = (const float*)d_in[12]; const float* enc_b2   = (const float*)d_in[13];
  const float* enc_mean_w = (const float*)d_in[14]; const float* enc_mean_b = (const float*)d_in[15];
  const float* enc_std_w  = (const float*)d_in[16]; const float* enc_std_b  = (const float*)d_in[17];
  const float* prior_w    = (const float*)d_in[18]; const float* prior_b    = (const float*)d_in[19];
  const float* prior_mean_w = (const float*)d_in[20]; const float* prior_mean_b = (const float*)d_in[21];
  const float* prior_std_w  = (const float*)d_in[22]; const float* prior_std_b  = (const float*)d_in[23];
  const float* dec_w1   = (const float*)d_in[24]; const float* dec_b1   = (const float*)d_in[25];
  const float* dec_w2   = (const float*)d_in[26]; const float* dec_b2   = (const float*)d_in[27];
  const float* dec_mean_w = (const float*)d_in[28]; const float* dec_mean_b = (const float*)d_in[29];
  const float* dec_std_w  = (const float*)d_in[30]; const float* dec_std_b  = (const float*)d_in[31];
  const float* gru_w_ih = (const float*)d_in[32];
  const float* gru_w_hh = (const float*)d_in[33];
  const float* fc_w = (const float*)d_in[34]; const float* fc_b = (const float*)d_in[35];

  float* out = (float*)d_out;
  float* logits = out + 2;                               // [T*B, V] flat (t-major)
  float* h_fin_out = out + 2 + (size_t)TT * BB * VV;     // [1, B, H]

  // ---- workspace bump allocator ----
  char* ws = (char*)d_ws;
  size_t off = 0;
  auto alloc = [&](size_t bytes) -> void* {
    off = (off + 255) & ~(size_t)255;
    void* p = ws + off;
    off += bytes;
    return p;
  };

  // bf16 weight copies; mean/std head pairs concatenated along N so each
  // head pair is one GEMM launch. Whole bf16 set (~41 MB + 32 MB fc) is
  // L2-resident across the 128-step scan.
  __bf16* b_phi_x_w1 = (__bf16*)alloc((size_t)HH * XX * 2);
  __bf16* b_phi_x_w2 = (__bf16*)alloc((size_t)HH * HH * 2);
  __bf16* b_phi_z_w  = (__bf16*)alloc((size_t)HH * ZZ * 2);
  __bf16* b_enc_w1   = (__bf16*)alloc((size_t)HH * 2 * HH * 2);
  __bf16* b_enc_w2   = (__bf16*)alloc((size_t)HH * HH * 2);
  __bf16* b_enc_head = (__bf16*)alloc((size_t)2 * ZZ * HH * 2);   // [mean;std]
  __bf16* b_prior    = (__bf16*)alloc((size_t)HH * HH * 2);
  __bf16* b_pri_head = (__bf16*)alloc((size_t)2 * ZZ * HH * 2);   // [mean;std]
  __bf16* b_dec_w1   = (__bf16*)alloc((size_t)HH * 2 * HH * 2);
  __bf16* b_dec_w2   = (__bf16*)alloc((size_t)HH * HH * 2);
  __bf16* b_dec_head = (__bf16*)alloc((size_t)2 * XX * HH * 2);   // [mean;std]
  __bf16* b_gru_ih   = (__bf16*)alloc((size_t)3 * HH * 2 * HH * 2);
  __bf16* b_gru_hh   = (__bf16*)alloc((size_t)3 * HH * HH * 2);
  __bf16* b_fc       = (__bf16*)alloc((size_t)VV * XX * 2);
  float* f_enc_head_b = (float*)alloc(2 * ZZ * 4);
  float* f_pri_head_b = (float*)alloc(2 * ZZ * 4);
  float* f_dec_head_b = (float*)alloc(2 * XX * 4);

  // fp32 activation buffers
  float* a1   = (float*)alloc(BB * HH * 4);
  float* phx  = (float*)alloc(BB * HH * 4);
  float* e1   = (float*)alloc(BB * HH * 4);
  float* ench = (float*)alloc(BB * HH * 4);
  float* pri  = (float*)alloc(BB * HH * 4);
  float* phz  = (float*)alloc(BB * HH * 4);
  float* dbuf1= (float*)alloc(BB * HH * 4);
  float* dech = (float*)alloc(BB * HH * 4);
  float* emes = (float*)alloc(BB * 2 * ZZ * 4);   // packed [em|es]
  float* pmps = (float*)alloc(BB * 2 * ZZ * 4);   // packed [pm|ps]
  float* dmds = (float*)alloc(BB * 2 * XX * 4);   // packed [dm|ds]
  float* zbuf = (float*)alloc(BB * ZZ * 4);
  float* gi   = (float*)alloc(BB * 3 * HH * 4);
  float* gh   = (float*)alloc(BB * 3 * HH * 4);
  float* hA   = (float*)alloc(BB * HH * 4);
  float* hB   = (float*)alloc(BB * HH * 4);
  float* preds= (float*)alloc((size_t)TT * BB * XX * 4);

  if (off > ws_size) return;  // workspace too small; nothing safe to do

  auto cvt = [&](const float* s, __bf16* d, int n) {
    cvt_bf16_kernel<<<(n + 255) / 256, 256, 0, stream>>>(s, d, n);
  };
  auto cpy = [&](const float* s, float* d, int n) {
    copyf_kernel<<<(n + 255) / 256, 256, 0, stream>>>(s, d, n);
  };

  // ---- one-time per call: zero accumulators, convert weights, init h ----
  zero2_kernel<<<1, 32, 0, stream>>>(out);
  cvt(phi_x_w1, b_phi_x_w1, HH * XX);
  cvt(phi_x_w2, b_phi_x_w2, HH * HH);
  cvt(phi_z_w,  b_phi_z_w,  HH * ZZ);
  cvt(enc_w1,   b_enc_w1,   HH * 2 * HH);
  cvt(enc_w2,   b_enc_w2,   HH * HH);
  cvt(enc_mean_w, b_enc_head,            ZZ * HH);
  cvt(enc_std_w,  b_enc_head + ZZ * HH,  ZZ * HH);
  cvt(prior_w,  b_prior,    HH * HH);
  cvt(prior_mean_w, b_pri_head,           ZZ * HH);
  cvt(prior_std_w,  b_pri_head + ZZ * HH, ZZ * HH);
  cvt(dec_w1,   b_dec_w1,   HH * 2 * HH);
  cvt(dec_w2,   b_dec_w2,   HH * HH);
  cvt(dec_mean_w, b_dec_head,            XX * HH);
  cvt(dec_std_w,  b_dec_head + XX * HH,  XX * HH);
  cvt(gru_w_ih, b_gru_ih,   3 * HH * 2 * HH);
  cvt(gru_w_hh, b_gru_hh,   3 * HH * HH);
  cvt(fc_w,     b_fc,       VV * XX);
  cpy(enc_mean_b, f_enc_head_b, ZZ);  cpy(enc_std_b, f_enc_head_b + ZZ, ZZ);
  cpy(prior_mean_b, f_pri_head_b, ZZ); cpy(prior_std_b, f_pri_head_b + ZZ, ZZ);
  cpy(dec_mean_b, f_dec_head_b, XX);  cpy(dec_std_b, f_dec_head_b + XX, XX);
  cpy(initial_state, hA, BB * HH);

  auto gemm1 = [&](const float* X1, int K1, int ldX1, const float* X2, int K2,
                   const __bf16* W, const float* bias, float* Y, int N,
                   int act1, int act2, int nsplit) {
    wmma_gemm<1><<<dim3(N / 64, 1), 128, 0, stream>>>(X1, K1, ldX1, X2, K2,
                                                      W, bias, Y, N,
                                                      act1, act2, nsplit);
  };

  // ---- recurrent scan (sequential; 13 GEMM + 5 aux launches per step) ----
  float* hprev = hA;
  float* hnext = hB;
  for (int t = 0; t < TT; ++t) {
    const float* x_t = x + (size_t)t * XX;  // row stride T*X -> x[:, t, :]

    gemm1(x_t, XX, TT * XX, nullptr, 0, b_phi_x_w1, phi_x_b1, a1, HH,
          ACT_RELU, ACT_RELU, HH);
    gemm1(a1, HH, HH, nullptr, 0, b_phi_x_w2, phi_x_b2, phx, HH,
          ACT_RELU, ACT_RELU, HH);

    gemm1(phx, HH, HH, hprev, HH, b_enc_w1, enc_b1, e1, HH,
          ACT_RELU, ACT_RELU, HH);                          // concat(phx,h)
    gemm1(e1, HH, HH, nullptr, 0, b_enc_w2, enc_b2, ench, HH,
          ACT_RELU, ACT_RELU, HH);
    gemm1(ench, HH, HH, nullptr, 0, b_enc_head, f_enc_head_b, emes, 2 * ZZ,
          ACT_NONE, ACT_SOFTPLUS, ZZ);                      // fused mean|std

    gemm1(hprev, HH, HH, nullptr, 0, b_prior, prior_b, pri, HH,
          ACT_RELU, ACT_RELU, HH);
    gemm1(pri, HH, HH, nullptr, 0, b_pri_head, f_pri_head_b, pmps, 2 * ZZ,
          ACT_NONE, ACT_SOFTPLUS, ZZ);                      // fused mean|std

    sample_z_kernel<<<(BB * ZZ + 255) / 256, 256, 0, stream>>>(emes, zbuf, t);
    gemm1(zbuf, ZZ, ZZ, nullptr, 0, b_phi_z_w, phi_z_b, phz, HH,
          ACT_RELU, ACT_RELU, HH);

    gemm1(phz, HH, HH, hprev, HH, b_dec_w1, dec_b1, dbuf1, HH,
          ACT_RELU, ACT_RELU, HH);                          // concat(phz,h)
    gemm1(dbuf1, HH, HH, nullptr, 0, b_dec_w2, dec_b2, dech, HH,
          ACT_RELU, ACT_RELU, HH);
    gemm1(dech, HH, HH, nullptr, 0, b_dec_head, f_dec_head_b, dmds, 2 * XX,
          ACT_SIGMOID, ACT_SOFTPLUS, XX);                   // fused mean|std

    pred_kernel<<<(BB * XX + 255) / 256, 256, 0, stream>>>(dmds, preds, t);
    kld_kernel<<<(BB * ZZ + 255) / 256, 256, 0, stream>>>(emes, pmps, out);
    nll_kernel<<<(BB * XX + 255) / 256, 256, 0, stream>>>(x_t, dmds, out + 1);

    gemm1(phx, HH, HH, phz, HH, b_gru_ih, nullptr, gi, 3 * HH,
          ACT_NONE, ACT_NONE, 3 * HH);                      // concat(phx,phz)
    gemm1(hprev, HH, HH, nullptr, 0, b_gru_hh, nullptr, gh, 3 * HH,
          ACT_NONE, ACT_NONE, 3 * HH);
    gru_kernel<<<(BB * HH + 255) / 256, 256, 0, stream>>>(gi, gh, hprev, hnext);

    float* tmp = hprev; hprev = hnext; hnext = tmp;
  }

  // ---- big fc GEMM: [4096 x 512] @ [512 x 32000] -> logits ----
  // MREP=4: each WG owns a 128x64 tile; B fragment reused by 8 WMMAs per
  // k-step -> fc weight L2 traffic 4 GB -> 1 GB, vs 512 MB HBM store floor.
  wmma_gemm<4><<<dim3(VV / 64, TT * BB / 128), 128, 0, stream>>>(
      preds, XX, XX, nullptr, 0, b_fc, fc_b, logits, VV,
      ACT_NONE, ACT_NONE, VV);

  // ---- final hidden state ----
  copyf_kernel<<<(BB * HH + 255) / 256, 256, 0, stream>>>(hprev, h_fin_out, BB * HH);
}
